// KerasSaSentTensorflow_41781441855620
// MI455X (gfx1250) — matrix operand order repaired and checked by reading, not compiled, not run on hardware
//
#include <hip/hip_runtime.h>
#include <math.h>

#define B_    32
#define T_    256
#define HID_  512
#define NG_   2048     // 4*HID
#define D0P_  352      // 350 padded to mult of 32
#define D1_   1024     // 2*HID

typedef __attribute__((ext_vector_type(16))) __bf16 v16bf;
typedef __attribute__((ext_vector_type(8)))  float  v8f;

struct BF16x16 { uint4 lo, hi; };

static __device__ __forceinline__ v8f wmma_bf16(v16bf a, v16bf b, v8f c) {
  return __builtin_amdgcn_wmma_f32_16x16x32_bf16(false, a, false, b, (short)0, c, false, false);
}

static __device__ __forceinline__ unsigned short f2bf(float f) {
  unsigned u = __float_as_uint(f);
  unsigned r = u + 0x7fffu + ((u >> 16) & 1u);   // round-to-nearest-even
  return (unsigned short)(r >> 16);
}

static __device__ __forceinline__ float sigm(float x) { return 1.f / (1.f + expf(-x)); }

// A-operand (16-bit, 16x32): lane = M, kh=lane>>4 ; halves at K = kh*8..+7 and 16+kh*8..+7
static __device__ __forceinline__ v16bf loadA_g(const unsigned short* base, int stride, int row0, int kt, int lane) {
  const unsigned short* p = base + (size_t)(row0 + (lane & 15)) * stride + kt * 32 + (lane >> 4) * 8;
  BF16x16 v; v.lo = *(const uint4*)p; v.hi = *(const uint4*)(p + 16);
  return __builtin_bit_cast(v16bf, v);
}
// B-operand (16-bit, 32x16): lane = N, kh=lane>>4 ; halves at K = kh*8..+7 and 16+kh*8..+7
static __device__ __forceinline__ v16bf loadB_g(const unsigned short* base, int stride, int row0, int kt, int lane) {
  const unsigned short* p = base + (size_t)(row0 + (lane & 15)) * stride + kt * 32 + (lane >> 4) * 8;
  BF16x16 v; v.lo = *(const uint4*)p; v.hi = *(const uint4*)(p + 16);
  return __builtin_bit_cast(v16bf, v);
}

// ---------------- embedding gather + concat + bf16 convert (pads 350->352) -------------
__global__ void embed_kernel(const int* __restrict__ ids, const int* __restrict__ msk,
                             const float* __restrict__ emb, const float* __restrict__ memb,
                             unsigned short* __restrict__ X) {
  int idx = blockIdx.x * blockDim.x + threadIdx.x;
  if (idx >= B_ * T_ * D0P_) return;
  int r = idx / D0P_, c = idx - r * D0P_;
  float v = 0.f;
  if (c < 300)      v = emb[(size_t)ids[r] * 300 + c];
  else if (c < 350) v = memb[(size_t)msk[r] * 50 + (c - 300)];
  X[idx] = f2bf(v);
}

// ---------------- weight transpose+convert: W (dx+512, 2048) -> WxT[2048][kxp], WhT[2048][512]
__global__ void wconv_kernel(const float* __restrict__ W, int dx, int kxp,
                             unsigned short* __restrict__ WxT, unsigned short* __restrict__ WhT) {
  int idx = blockIdx.x * blockDim.x + threadIdx.x;
  int totX = NG_ * kxp;
  if (idx < totX) {
    int n = idx / kxp, k = idx - n * kxp;
    WxT[idx] = f2bf(k < dx ? W[(size_t)k * NG_ + n] : 0.f);
  } else {
    int e = idx - totX;
    if (e >= NG_ * HID_) return;
    int n = e / HID_, k = e - n * HID_;
    WhT[e] = f2bf(W[(size_t)(dx + k) * NG_ + n]);
  }
}

// ---------------- batch input GEMM: pre[dir][t][b][2048] = X @ Wx + bias ----------------
// X: [8192][K] bf16 row-major (row = b*T + t); WT: [2048][K] bf16 (= Wx^T)
__global__ __launch_bounds__(128) void gemm_pre_kernel(
    const unsigned short* __restrict__ X,
    const unsigned short* __restrict__ WT0, const unsigned short* __restrict__ WT1,
    const float* __restrict__ bias0, const float* __restrict__ bias1,
    float* __restrict__ pre, int K) {
  int dir = blockIdx.y;
  const unsigned short* WT = dir ? WT1 : WT0;
  const float* bias = dir ? bias1 : bias0;
  float* preD = pre + (size_t)dir * T_ * B_ * NG_;

  int lane = threadIdx.x & 31;
  int wv   = threadIdx.x >> 5;
  int grp  = blockIdx.x * 4 + wv;         // 16384 groups = 512 Mtiles * 32 Ngroups
  int mt   = grp >> 5;                    // 0..511
  int n0   = (grp & 31) * 64;             // 4 N-tiles per wave

  v8f acc[4] = {};
  int nkt = K >> 5;
  for (int kt = 0; kt < nkt; ++kt) {
    v16bf a = loadA_g(X, K, mt * 16, kt, lane);
#pragma unroll
    for (int q = 0; q < 4; ++q) {
      v16bf b = loadB_g(WT, K, n0 + q * 16, kt, lane);
      acc[q] = wmma_bf16(a, b, acc[q]);
    }
  }
#pragma unroll
  for (int q = 0; q < 4; ++q) {
    int N = n0 + q * 16 + (lane & 15);
    float bv = bias[N];
#pragma unroll
    for (int r = 0; r < 8; ++r) {
      int M = mt * 16 + ((lane < 16) ? r : r + 8);
      int b = M >> 8;        // row = b*T + t, T = 256
      int t = M & 255;
      preD[((size_t)t * B_ + b) * NG_ + N] = acc[q][r] + bv;
    }
  }
}

// ---------------- persistent LSTM scan (one block per direction) ------------------------
// WhT: [2048][512] bf16 per direction; pre: [dir][T][B][2048] f32; out: [B][T][1024] f32
__global__ __launch_bounds__(1024) void lstm_scan_kernel(
    const unsigned short* __restrict__ WhF, const unsigned short* __restrict__ WhB,
    const float* __restrict__ pre, const int* __restrict__ length,
    float* __restrict__ out) {
  int dir = blockIdx.x;
  const unsigned short* Wh = dir ? WhB : WhF;
  const float* preD = pre + (size_t)dir * T_ * B_ * NG_;

  __shared__ __align__(16) unsigned short hbuf[2][B_][HID_];
  __shared__ int lenS[B_];

  int tid = threadIdx.x;
  for (int i = tid; i < 2 * B_ * HID_; i += 1024) ((unsigned short*)hbuf)[i] = 0;
  if (tid < B_) lenS[tid] = length[tid];
  __syncthreads();

  int lane = tid & 31;
  int ht   = tid >> 5;                    // hidden tile 0..31
  int col  = ht * 16 + (lane & 15);

  v8f creg[2] = {}, hreg[2] = {};

  for (int s = 0; s < T_; ++s) {
    int cur = s & 1, nxt = cur ^ 1;
#pragma unroll
    for (int mt = 0; mt < 2; ++mt) {
      v8f ai = {}, aj = {}, af = {}, ao = {};
      for (int kt = 0; kt < HID_ / 32; ++kt) {
        const unsigned short* p = &hbuf[cur][mt * 16 + (lane & 15)][kt * 32 + (lane >> 4) * 8];
        BF16x16 av; av.lo = *(const uint4*)p; av.hi = *(const uint4*)(p + 16);
        v16bf a = __builtin_bit_cast(v16bf, av);
        v16bf b0 = loadB_g(Wh, HID_, 0 * HID_ + ht * 16, kt, lane); ai = wmma_bf16(a, b0, ai);
        v16bf b1 = loadB_g(Wh, HID_, 1 * HID_ + ht * 16, kt, lane); aj = wmma_bf16(a, b1, aj);
        v16bf b2 = loadB_g(Wh, HID_, 2 * HID_ + ht * 16, kt, lane); af = wmma_bf16(a, b2, af);
        v16bf b3 = loadB_g(Wh, HID_, 3 * HID_ + ht * 16, kt, lane); ao = wmma_bf16(a, b3, ao);
      }
#pragma unroll
      for (int r = 0; r < 8; ++r) {
        int b   = mt * 16 + ((lane < 16) ? r : r + 8);
        int len = lenS[b];
        int t   = (dir == 0) ? s : ((s < len) ? (len - 1 - s) : s);
        size_t prow = ((size_t)t * B_ + b) * NG_;
        float zi = ai[r] + preD[prow + col];
        float zj = aj[r] + preD[prow + HID_ + col];
        float zf = af[r] + preD[prow + 2 * HID_ + col];
        float zo = ao[r] + preD[prow + 3 * HID_ + col];
        float cv = creg[mt][r], hv = hreg[mt][r];
        float cn = sigm(zf + 1.f) * cv + sigm(zi) * tanhf(zj);
        float hn = sigm(zo) * tanhf(cn);
        bool m = s < len;
        cv = m ? cn : cv;
        hv = m ? hn : hv;
        creg[mt][r] = cv;
        hreg[mt][r] = hv;
        hbuf[nxt][b][col] = f2bf(hv);
        out[((size_t)b * T_ + t) * D1_ + dir * HID_ + col] = m ? hn : 0.f;
      }
    }
    __syncthreads();
  }
}

// ---------------- f32 -> bf16 elementwise --------------------------------------------
__global__ void cvt_bf16_kernel(const float* __restrict__ in, unsigned short* __restrict__ o, int n) {
  int i = blockIdx.x * blockDim.x + threadIdx.x;
  if (i < n) o[i] = f2bf(in[i]);
}

// ---------------- CRF emissions: e[b][t][2] = out1[b][t] @ crf_w + crf_b ---------------
__global__ void crf_e_kernel(const float* __restrict__ o1, const float* __restrict__ cw,
                             const float* __restrict__ cb, float* __restrict__ e) {
  int i = blockIdx.x * blockDim.x + threadIdx.x;
  if (i >= B_ * T_) return;
  const float* row = o1 + (size_t)i * D1_;
  float s0 = cb[0], s1 = cb[1];
  for (int k = 0; k < D1_; ++k) { float v = row[k]; s0 += v * cw[2 * k]; s1 += v * cw[2 * k + 1]; }
  e[2 * i] = s0; e[2 * i + 1] = s1;
}

// ---------------- CRF forward scan, emit p(class=1) per (b,t) --------------------------
__global__ void crf_scan_kernel(const float* __restrict__ e, const float* __restrict__ tr,
                                const int* __restrict__ length, float* __restrict__ p1) {
  int b = threadIdx.x;
  if (b >= B_) return;
  int len = length[b];
  float t00 = tr[0], t01 = tr[1], t10 = tr[2], t11 = tr[3];
  float a0 = e[(size_t)(b * T_) * 2], a1 = e[(size_t)(b * T_) * 2 + 1];
  {
    float mx = fmaxf(a0, a1);
    float z = expf(a0 - mx) + expf(a1 - mx);
    p1[b * T_] = (len > 0) ? expf(a1 - mx) / z : 0.f;
  }
  for (int t = 1; t < T_; ++t) {
    if (t < len) {
      float x00 = a0 + t00, x10 = a1 + t10;
      float m0 = fmaxf(x00, x10);
      float n0 = m0 + logf(expf(x00 - m0) + expf(x10 - m0)) + e[(size_t)(b * T_ + t) * 2];
      float x01 = a0 + t01, x11 = a1 + t11;
      float m1 = fmaxf(x01, x11);
      float n1 = m1 + logf(expf(x01 - m1) + expf(x11 - m1)) + e[(size_t)(b * T_ + t) * 2 + 1];
      a0 = n0; a1 = n1;
      float mx = fmaxf(a0, a1);
      float z = expf(a0 - mx) + expf(a1 - mx);
      p1[b * T_ + t] = expf(a1 - mx) / z;
    } else {
      p1[b * T_ + t] = 0.f;
    }
  }
}

// ---------------- weighted pooling: sv[b][k] = sum_t p1[b][t] * out1[b][t][k] ----------
__global__ __launch_bounds__(1024) void sv_kernel(const float* __restrict__ o1,
                                                  const float* __restrict__ p1,
                                                  float* __restrict__ sv) {
  int b = blockIdx.x, k = threadIdx.x;
  float s = 0.f;
  for (int t = 0; t < T_; ++t) s += p1[b * T_ + t] * o1[((size_t)b * T_ + t) * D1_ + k];
  sv[b * D1_ + k] = s;
}

// ---------------- final logits + softmax ----------------------------------------------
__global__ void logits_kernel(const float* __restrict__ sv, const float* __restrict__ lw,
                              const float* __restrict__ lb, float* __restrict__ o) {
  int b = threadIdx.x;
  if (b >= B_) return;
  float l0 = lb[0], l1 = lb[1], l2 = lb[2];
  for (int k = 0; k < D1_; ++k) {
    float v = sv[b * D1_ + k];
    l0 += v * lw[3 * k]; l1 += v * lw[3 * k + 1]; l2 += v * lw[3 * k + 2];
  }
  float mx = fmaxf(l0, fmaxf(l1, l2));
  float e0 = expf(l0 - mx), e1 = expf(l1 - mx), e2 = expf(l2 - mx);
  float z = e0 + e1 + e2;
  o[b * 3] = e0 / z; o[b * 3 + 1] = e1 / z; o[b * 3 + 2] = e2 / z;
}

extern "C" void kernel_launch(void* const* d_in, const int* in_sizes, int n_in,
                              void* d_out, int out_size, void* d_ws, size_t ws_size,
                              hipStream_t stream) {
  (void)in_sizes; (void)n_in; (void)out_size; (void)ws_size;
  const int*   ids   = (const int*)d_in[0];
  const int*   msk   = (const int*)d_in[1];
  const int*   len   = (const int*)d_in[2];
  const float* emb   = (const float*)d_in[3];
  const float* memb  = (const float*)d_in[4];
  const float* trans = (const float*)d_in[5];
  const float* w_fw0 = (const float*)d_in[6];
  const float* b_fw0 = (const float*)d_in[7];
  const float* w_bw0 = (const float*)d_in[8];
  const float* b_bw0 = (const float*)d_in[9];
  const float* w_fw1 = (const float*)d_in[10];
  const float* b_fw1 = (const float*)d_in[11];
  const float* w_bw1 = (const float*)d_in[12];
  const float* b_bw1 = (const float*)d_in[13];
  const float* crf_w = (const float*)d_in[14];
  const float* crf_b = (const float*)d_in[15];
  const float* lg_w  = (const float*)d_in[16];
  const float* lg_b  = (const float*)d_in[17];

  char* wsp = (char*)d_ws;
  size_t off = 0;
  auto carve = [&](size_t bytes) -> char* {
    char* p = wsp + off;
    off = (off + bytes + 255) & ~(size_t)255;
    return p;
  };
  unsigned short* X0    = (unsigned short*)carve((size_t)B_ * T_ * D0P_ * 2);
  unsigned short* X1    = (unsigned short*)carve((size_t)B_ * T_ * D1_ * 2);
  unsigned short* WxT0f = (unsigned short*)carve((size_t)NG_ * D0P_ * 2);
  unsigned short* WxT0b = (unsigned short*)carve((size_t)NG_ * D0P_ * 2);
  unsigned short* WhT0f = (unsigned short*)carve((size_t)NG_ * HID_ * 2);
  unsigned short* WhT0b = (unsigned short*)carve((size_t)NG_ * HID_ * 2);
  unsigned short* WxT1f = (unsigned short*)carve((size_t)NG_ * D1_ * 2);
  unsigned short* WxT1b = (unsigned short*)carve((size_t)NG_ * D1_ * 2);
  unsigned short* WhT1f = (unsigned short*)carve((size_t)NG_ * HID_ * 2);
  unsigned short* WhT1b = (unsigned short*)carve((size_t)NG_ * HID_ * 2);
  float* pre  = (float*)carve((size_t)2 * T_ * B_ * NG_ * 4);
  float* out0 = (float*)carve((size_t)B_ * T_ * D1_ * 4);
  float* out1 = (float*)carve((size_t)B_ * T_ * D1_ * 4);
  float* eb   = (float*)carve((size_t)B_ * T_ * 2 * 4);
  float* p1   = (float*)carve((size_t)B_ * T_ * 4);
  float* sv   = (float*)carve((size_t)B_ * D1_ * 4);

  // 1) embedding gather + concat -> bf16
  {
    int n = B_ * T_ * D0P_;
    embed_kernel<<<(n + 255) / 256, 256, 0, stream>>>(ids, msk, emb, memb, X0);
  }
  // 2) weight transpose/convert
  {
    int n0 = NG_ * (D0P_ + HID_);
    wconv_kernel<<<(n0 + 255) / 256, 256, 0, stream>>>(w_fw0, 350, D0P_, WxT0f, WhT0f);
    wconv_kernel<<<(n0 + 255) / 256, 256, 0, stream>>>(w_bw0, 350, D0P_, WxT0b, WhT0b);
    int n1 = NG_ * (D1_ + HID_);
    wconv_kernel<<<(n1 + 255) / 256, 256, 0, stream>>>(w_fw1, D1_, D1_, WxT1f, WhT1f);
    wconv_kernel<<<(n1 + 255) / 256, 256, 0, stream>>>(w_bw1, D1_, D1_, WxT1b, WhT1b);
  }
  // 3) layer-0 input GEMM (both directions), 4) layer-0 scan
  {
    dim3 g(4096, 2);
    gemm_pre_kernel<<<g, 128, 0, stream>>>(X0, WxT0f, WxT0b, b_fw0, b_bw0, pre, D0P_);
    lstm_scan_kernel<<<2, 1024, 0, stream>>>(WhT0f, WhT0b, pre, len, out0);
  }
  // 5) out0 -> bf16, 6) layer-1 input GEMM, 7) layer-1 scan
  {
    int n = B_ * T_ * D1_;
    cvt_bf16_kernel<<<(n + 255) / 256, 256, 0, stream>>>(out0, X1, n);
    dim3 g(4096, 2);
    gemm_pre_kernel<<<g, 128, 0, stream>>>(X1, WxT1f, WxT1b, b_fw1, b_bw1, pre, D1_);
    lstm_scan_kernel<<<2, 1024, 0, stream>>>(WhT1f, WhT1b, pre, len, out1);
  }
  // 8) CRF emissions, 9) CRF scan, 10) pooling, 11) logits
  crf_e_kernel<<<(B_ * T_ + 255) / 256, 256, 0, stream>>>(out1, crf_w, crf_b, eb);
  crf_scan_kernel<<<1, 32, 0, stream>>>(eb, trans, len, p1);
  sv_kernel<<<B_, 1024, 0, stream>>>(out1, p1, sv);
  logits_kernel<<<1, 32, 0, stream>>>(sv, lg_w, lg_b, (float*)d_out);
}
